// Decoder1_17738214932668
// MI455X (gfx1250) — matrix operand order repaired and checked by reading, not compiled, not run on hardware
//
#include <hip/hip_runtime.h>

typedef __attribute__((ext_vector_type(16))) _Float16 v16h;
typedef __attribute__((ext_vector_type(8)))  float    v8f;
typedef __attribute__((ext_vector_type(4)))  unsigned uv4;

#define EPSF 1e-5f

union Frag16 { v16h h; uv4 q[2]; };

// ---------------- elementwise helpers ----------------

__global__ __launch_bounds__(256) void k_f32tof16(const float* __restrict__ in,
                                                  _Float16* __restrict__ out, long total) {
  long stride = (long)gridDim.x * 256;
  for (long i = (long)blockIdx.x * 256 + threadIdx.x; i < total; i += stride)
    out[i] = (_Float16)in[i];
}

// w[co][ci][kh][kw] (f32) -> wp[co][t][ci] (f16), t = kh*3+kw
__global__ __launch_bounds__(256) void k_repack_w(const float* __restrict__ w,
                                                  _Float16* __restrict__ wp,
                                                  int Cout, int Cin) {
  long total = (long)Cout * Cin * 9;
  long stride = (long)gridDim.x * 256;
  for (long i = (long)blockIdx.x * 256 + threadIdx.x; i < total; i += stride) {
    int ci = (int)(i % Cin);
    long r = i / Cin;
    int t  = (int)(r % 9);
    int co = (int)(r / 9);
    wp[i] = (_Float16)w[((long)co * Cin + ci) * 9 + t];
  }
}

// ---------------- implicit-GEMM 3x3 conv + bias + ReLU (reflect pad 1) ----------------
// 8 waves / block. Wave computes 16(Cout) x 32(pixels): one A fragment feeds two
// B fragments / two accumulators. LDS double-buffered xs[2][3][34 cols][32 ch] f16.
__global__ __launch_bounds__(256) void k_conv3x3_wmma(
    const _Float16* __restrict__ x,   // [N,Cin,H,W]
    const _Float16* __restrict__ wp,  // [Cout,9,Cin]
    const float*    __restrict__ bias,// [Cout]
    _Float16*       __restrict__ y,   // [N,Cout,H,W]
    int Cin, int Cout, int H, int W)
{
  const int lane = threadIdx.x & 31;
  const int wid  = threadIdx.x >> 5;
  const int tilesPerRow = W >> 5;
  const int n  = blockIdx.z;
  const int h  = blockIdx.x / tilesPerRow;
  const int w0 = (blockIdx.x - h * tilesPerRow) << 5;
  const int co0 = (blockIdx.y * 8 + wid) << 4;
  const bool active = (co0 < Cout);

  __shared__ __align__(16) _Float16 xs[2][3][34][32];

  const int mrow = lane & 15;                 // A: cout row; B: pixel column
  const int kb_a = (lane < 16) ? 0 : 16;      // contiguous k = 16h + s (matches B)
  const int kb_b = (lane < 16) ? 0 : 16;      // contiguous k = 16h + s (matches A)

  int rh[3];
  rh[0] = (h == 0) ? 1 : (h - 1);
  rh[1] = h;
  rh[2] = (h == H - 1) ? (H - 2) : (h + 1);
  const int gl = (w0 == 0) ? 1 : (w0 - 1);                 // reflected left col
  const int gr = (w0 + 32 >= W) ? (W - 2) : (w0 + 32);     // reflected right col

  // stage one 32-channel slab; each thread owns (row, channel-pair), packs two
  // f16 channels into one u32 LDS store per column.
  auto stage = [&](int ci0, int buf) {
    for (int idx = threadIdx.x; idx < 48; idx += 256) {
      int r  = idx / 16;
      int ci = (idx % 16) * 2;
      const _Float16* s0 = x + (((long)n * Cin + ci0 + ci)     * H + rh[r]) * W;
      const _Float16* s1 = x + (((long)n * Cin + ci0 + ci + 1) * H + rh[r]) * W;
      unsigned lo, hi;
      lo = *(const unsigned short*)(s0 + gl);
      hi = *(const unsigned short*)(s1 + gl);
      *(unsigned*)&xs[buf][r][0][ci] = lo | (hi << 16);
      lo = *(const unsigned short*)(s0 + gr);
      hi = *(const unsigned short*)(s1 + gr);
      *(unsigned*)&xs[buf][r][33][ci] = lo | (hi << 16);
#pragma unroll 8
      for (int c = 1; c < 33; ++c) {
        lo = *(const unsigned short*)(s0 + w0 - 1 + c);
        hi = *(const unsigned short*)(s1 + w0 - 1 + c);
        *(unsigned*)&xs[buf][r][c][ci] = lo | (hi << 16);
      }
    }
  };

  v8f acc0 = {};
  v8f acc1 = {};

  auto compute = [&](int ch) {
    if (!active) return;
    const int buf = ch & 1;
    const int ci0 = ch << 5;
    const _Float16* wbase = wp + ((long)(co0 + mrow) * 9) * Cin + ci0;
#pragma unroll
    for (int t = 0; t < 9; ++t) {
      const int kh = t / 3, kw = t - kh * 3;
      Frag16 a, b0, b1;
      const _Float16* wrow  = wbase + t * Cin;
      const _Float16* brow0 = &xs[buf][kh][mrow + kw][0];
      const _Float16* brow1 = &xs[buf][kh][mrow + 16 + kw][0];
      a.q[0]  = *(const uv4*)(wrow + kb_a);        // 16B spans, 16B aligned
      a.q[1]  = *(const uv4*)(wrow + kb_a + 8);
      b0.q[0] = *(const uv4*)(brow0 + kb_b);
      b0.q[1] = *(const uv4*)(brow0 + kb_b + 8);
      b1.q[0] = *(const uv4*)(brow1 + kb_b);
      b1.q[1] = *(const uv4*)(brow1 + kb_b + 8);
      acc0 = __builtin_amdgcn_wmma_f32_16x16x32_f16(
          false, a.h, false, b0.h, (short)0, acc0, false, false);
      acc1 = __builtin_amdgcn_wmma_f32_16x16x32_f16(
          false, a.h, false, b1.h, (short)0, acc1, false, false);
    }
  };

  const int nchunk = Cin >> 5;   // >= 4 for all layers
  stage(0, 0);
  __syncthreads();
  for (int ch = 0; ch < nchunk - 1; ++ch) {
    stage((ch + 1) << 5, (ch + 1) & 1);   // overlap next slab with current compute
    compute(ch);
    __syncthreads();
  }
  compute(nchunk - 1);

  if (active) {
#pragma unroll
    for (int r = 0; r < 8; ++r) {
      const int m  = (lane < 16) ? r : (r + 8);   // C/D layout: VGPR r -> M=r / r+8
      const int co = co0 + m;
      const float bb = bias[co];
      long o = (((long)n * Cout + co) * H + h) * W + w0 + mrow;
      float v0 = acc0[r] + bb;
      float v1 = acc1[r] + bb;
      y[o]      = (_Float16)(v0 > 0.f ? v0 : 0.f);
      y[o + 16] = (_Float16)(v1 > 0.f ? v1 : 0.f);
    }
  }
}

// ---------------- wavelet unpool fused with rpad-crop shift ----------------
__global__ __launch_bounds__(256) void k_unpool_shift(
    const _Float16* __restrict__ ll, const float* __restrict__ lh,
    const float* __restrict__ hl, const float* __restrict__ hh,
    _Float16* __restrict__ out, int NC, int Hin, int Win)
{
  const int Ho = Hin * 2, Wo = Win * 2;
  long total = (long)NC * Ho * Wo;
  long stride = (long)gridDim.x * 256;
  for (long idx = (long)blockIdx.x * 256 + threadIdx.x; idx < total; idx += stride) {
    int j = (int)(idx % Wo);
    long r = idx / Wo;
    int i = (int)(r % Ho);
    long nc = r / Ho;
    int si = (i == 0) ? 1 : (i - 1);
    int sj = (j == 0) ? 1 : (j - 1);
    int iu = si >> 1, aa = si & 1, ju = sj >> 1, bb = sj & 1;
    long base = (nc * Hin + iu) * Win + ju;
    float l  = (float)ll[base];
    float sb = bb ? 1.f : -1.f;
    float sa = aa ? 1.f : -1.f;
    out[idx] = (_Float16)(0.5f * (l + sb * lh[base] + sa * hl[base] + sa * sb * hh[base]));
  }
}

// final unpool + full reflect pad -> f32 output [NC][2H+2][2W+2]
__global__ __launch_bounds__(256) void k_unpool_rpad(
    const _Float16* __restrict__ ll, const float* __restrict__ lh,
    const float* __restrict__ hl, const float* __restrict__ hh,
    float* __restrict__ out, int NC, int Hin, int Win)
{
  const int Hu = Hin * 2, Wu = Win * 2, Ho = Hu + 2, Wo = Wu + 2;
  long total = (long)NC * Ho * Wo;
  long stride = (long)gridDim.x * 256;
  for (long idx = (long)blockIdx.x * 256 + threadIdx.x; idx < total; idx += stride) {
    int j = (int)(idx % Wo);
    long r = idx / Wo;
    int i = (int)(r % Ho);
    long nc = r / Ho;
    int pi = (i == 0) ? 1 : ((i == Ho - 1) ? (Hu - 2) : (i - 1));
    int pj = (j == 0) ? 1 : ((j == Wo - 1) ? (Wu - 2) : (j - 1));
    int iu = pi >> 1, aa = pi & 1, ju = pj >> 1, bb = pj & 1;
    long base = (nc * Hin + iu) * Win + ju;
    float l  = (float)ll[base];
    float sb = bb ? 1.f : -1.f;
    float sa = aa ? 1.f : -1.f;
    out[idx] = 0.5f * (l + sb * lh[base] + sa * hl[base] + sa * sb * hh[base]);
  }
}

// ---------------- AdaIN pieces ----------------

__global__ __launch_bounds__(256) void k_pool_masks(const float* __restrict__ m,
                                                    float* __restrict__ pm) {
  long total = 8L * 128 * 128;
  long stride = (long)gridDim.x * 256;
  for (long idx = (long)blockIdx.x * 256 + threadIdx.x; idx < total; idx += stride) {
    int w = (int)(idx % 128);
    long t = idx / 128;
    int h = (int)(t % 128);
    int k = (int)(t / 128);
    pm[idx] = m[((long)k * 256 + 2 * h) * 256 + 2 * w];
  }
}

__global__ __launch_bounds__(256) void k_mask_flags(const float* __restrict__ cm,
                                                    const float* __restrict__ sm,
                                                    float* __restrict__ flags,
                                                    float* __restrict__ sempty) {
  const int k = blockIdx.x;
  const int HW = 256 * 256;
  float mc = 0.f, ms = 0.f;
  for (int p = threadIdx.x; p < HW; p += 256) {
    mc = fmaxf(mc, cm[(long)k * HW + p]);
    ms = fmaxf(ms, sm[(long)k * HW + p]);
  }
  for (int o = 16; o > 0; o >>= 1) {
    mc = fmaxf(mc, __shfl_down(mc, o));
    ms = fmaxf(ms, __shfl_down(ms, o));
  }
  __shared__ float rc[8], rs[8];
  int lane = threadIdx.x & 31, wid = threadIdx.x >> 5;
  if (lane == 0) { rc[wid] = mc; rs[wid] = ms; }
  __syncthreads();
  if (threadIdx.x == 0) {
    float a = 0.f, b = 0.f;
    for (int w = 0; w < 8; ++w) { a = fmaxf(a, rc[w]); b = fmaxf(b, rs[w]); }
    flags[k]  = (a != 0.f) ? 1.f : 0.f;
    sempty[k] = (b == 0.f) ? 1.f : 0.f;
  }
}

// masked (+optional global) stats for feat+EPS. One block per (n,c). feat16 xor feat32.
__global__ __launch_bounds__(256) void k_stats(
    const _Float16* __restrict__ feat16, const float* __restrict__ feat32,
    const float* __restrict__ pmask,      // [8][HW]
    float* __restrict__ out_mean, float* __restrict__ out_std,  // [(k*N+n)*C+c]
    float* __restrict__ g_mean, float* __restrict__ g_std,      // [n*C+c] or null
    int N, int C, int HW)
{
  const int nc = blockIdx.x;
  const int n = nc / C, c = nc % C;
  const int tid = threadIdx.x, lane = tid & 31, wid = tid >> 5;
  float cnt[8], s1[8], s2[8];
#pragma unroll
  for (int k = 0; k < 8; ++k) { cnt[k] = 0.f; s1[k] = 0.f; s2[k] = 0.f; }
  float gc = 0.f, g1 = 0.f, g2 = 0.f;
  const long base = (long)nc * HW;
  for (int p = tid; p < HW; p += 256) {
    float f = (feat16 ? (float)feat16[base + p] : feat32[base + p]) + EPSF;
    float nz = (f != 0.f) ? 1.f : 0.f;
    float ff = f * f;
    if (g_mean) { gc += nz; g1 += f; g2 += ff; }
#pragma unroll
    for (int k = 0; k < 8; ++k) {
      float m = pmask[(long)k * HW + p];
      cnt[k] += m * nz; s1[k] += m * f; s2[k] += m * ff;
    }
  }
#pragma unroll
  for (int k = 0; k < 8; ++k)
    for (int o = 16; o > 0; o >>= 1) {
      cnt[k] += __shfl_down(cnt[k], o);
      s1[k]  += __shfl_down(s1[k], o);
      s2[k]  += __shfl_down(s2[k], o);
    }
  for (int o = 16; o > 0; o >>= 1) {
    gc += __shfl_down(gc, o); g1 += __shfl_down(g1, o); g2 += __shfl_down(g2, o);
  }
  __shared__ float red[8][27];
  if (lane == 0) {
#pragma unroll
    for (int k = 0; k < 8; ++k) {
      red[wid][k * 3 + 0] = cnt[k];
      red[wid][k * 3 + 1] = s1[k];
      red[wid][k * 3 + 2] = s2[k];
    }
    red[wid][24] = gc; red[wid][25] = g1; red[wid][26] = g2;
  }
  __syncthreads();
  if (tid < 8) {
    int k = tid;
    float C0 = 0.f, C1 = 0.f, C2 = 0.f;
    for (int w = 0; w < 8; ++w) {
      C0 += red[w][k * 3 + 0]; C1 += red[w][k * 3 + 1]; C2 += red[w][k * 3 + 2];
    }
    float mean = C1 / fmaxf(C0, 1.f);
    float var  = (C2 - C0 * mean * mean) / fmaxf(C0 - 1.f, 1.f) + EPSF;
    int oi = (k * N + n) * C + c;
    out_mean[oi] = mean;
    out_std[oi]  = sqrtf(var);
  }
  if (g_mean && tid == 8) {
    float C0 = 0.f, C1 = 0.f, C2 = 0.f;
    for (int w = 0; w < 8; ++w) { C0 += red[w][24]; C1 += red[w][25]; C2 += red[w][26]; }
    float mean = C1 / fmaxf(C0, 1.f);
    float var  = (C2 - C0 * mean * mean) / fmaxf(C0 - 1.f, 1.f) + EPSF;
    g_mean[nc] = mean;
    g_std[nc]  = sqrtf(var);
  }
}

__global__ __launch_bounds__(256) void k_scale_shift(
    const float* __restrict__ c_mean, const float* __restrict__ c_std,
    const float* __restrict__ sm_mean, const float* __restrict__ sm_std,
    const float* __restrict__ sg_mean, const float* __restrict__ sg_std,
    const float* __restrict__ flags, const float* __restrict__ sempty,
    float* __restrict__ scale, float* __restrict__ shift, int N, int C)
{
  int idx = blockIdx.x * 256 + threadIdx.x;
  int total = 8 * N * C;
  if (idx >= total) return;
  int c = idx % C;
  int r = idx / C;
  int n = r % N;
  int k = r / N;
  float cs = c_std[idx], cmn = c_mean[idx];
  bool emp = (sempty[k] != 0.f);
  float sm = emp ? sg_mean[n * C + c] : sm_mean[idx];
  float ss = emp ? sg_std[n * C + c]  : sm_std[idx];
  float f = flags[k];
  scale[idx] = f * ss / cs;
  shift[idx] = f * (sm - cmn * ss / cs);
}

__global__ __launch_bounds__(256) void k_adain_apply(
    const _Float16* __restrict__ content, const float* __restrict__ pm,
    const float* __restrict__ scale, const float* __restrict__ shift,
    _Float16* __restrict__ out, int N, int C, int HW)
{
  long total = (long)N * C * HW;
  long stride = (long)gridDim.x * 256;
  for (long idx = (long)blockIdx.x * 256 + threadIdx.x; idx < total; idx += stride) {
    int p = (int)(idx % HW);
    long r = idx / HW;
    int c = (int)(r % C);
    int n = (int)(r / C);
    float f = (float)content[idx] + EPSF;
    float S = 0.f, T = 0.f;
#pragma unroll
    for (int k = 0; k < 8; ++k) {
      float m = pm[(long)k * HW + p];
      int si = (k * N + n) * C + c;
      S += m * scale[si];
      T += m * shift[si];
    }
    out[idx] = (_Float16)(f * S + T);
  }
}

// ---------------- launch ----------------

extern "C" void kernel_launch(void* const* d_in, const int* in_sizes, int n_in,
                              void* d_out, int out_size, void* d_ws, size_t ws_size,
                              hipStream_t stream) {
  const float* f1    = (const float*)d_in[0];
  const float* f2    = (const float*)d_in[1];
  const float* s2lh  = (const float*)d_in[2];
  const float* s2hl  = (const float*)d_in[3];
  const float* s2hh  = (const float*)d_in[4];
  const float* s1lh  = (const float*)d_in[5];
  const float* s1hl  = (const float*)d_in[6];
  const float* s1hh  = (const float*)d_in[7];
  const float* s0lh  = (const float*)d_in[8];
  const float* s0hl  = (const float*)d_in[9];
  const float* s0hh  = (const float*)d_in[10];
  const float* cmask = (const float*)d_in[11];
  const float* smask = (const float*)d_in[12];
  const float* W11 = (const float*)d_in[13]; const float* B11 = (const float*)d_in[14];
  const float* W12 = (const float*)d_in[15]; const float* B12 = (const float*)d_in[16];
  const float* W21 = (const float*)d_in[17]; const float* B21 = (const float*)d_in[18];
  const float* W22 = (const float*)d_in[19]; const float* B22 = (const float*)d_in[20];
  const float* W23 = (const float*)d_in[21]; const float* B23 = (const float*)d_in[22];
  const float* W24 = (const float*)d_in[23]; const float* B24 = (const float*)d_in[24];
  const float* W31 = (const float*)d_in[25]; const float* B31 = (const float*)d_in[26];
  const float* W32 = (const float*)d_in[27]; const float* B32 = (const float*)d_in[28];
  const float* W33 = (const float*)d_in[29]; const float* B33 = (const float*)d_in[30];

  char* ws = (char*)d_ws;
  size_t off = 0;
  auto alloc = [&](size_t bytes) -> void* {
    void* p = (void*)(ws + off);
    off += (bytes + 255) & ~(size_t)255;
    return p;
  };

  _Float16* wp11 = (_Float16*)alloc((size_t)512 * 512 * 9 * 2);
  _Float16* wp12 = (_Float16*)alloc((size_t)256 * 512 * 9 * 2);
  _Float16* wp21 = (_Float16*)alloc((size_t)256 * 256 * 9 * 2);
  _Float16* wp22 = (_Float16*)alloc((size_t)256 * 256 * 9 * 2);
  _Float16* wp23 = (_Float16*)alloc((size_t)256 * 256 * 9 * 2);
  _Float16* wp24 = (_Float16*)alloc((size_t)128 * 256 * 9 * 2);
  _Float16* wp31 = (_Float16*)alloc((size_t)128 * 128 * 9 * 2);
  _Float16* wp32 = (_Float16*)alloc((size_t)128 * 128 * 9 * 2);
  _Float16* wp33 = (_Float16*)alloc((size_t)64  * 128 * 9 * 2);
  _Float16* bufA = (_Float16*)alloc((size_t)4 * 128 * 128 * 128 * 2);
  _Float16* bufB = (_Float16*)alloc((size_t)4 * 128 * 128 * 128 * 2);
  float* cmp    = (float*)alloc((size_t)8 * 128 * 128 * 4);
  float* smp    = (float*)alloc((size_t)8 * 128 * 128 * 4);
  float* flags  = (float*)alloc(8 * 4);
  float* sempty = (float*)alloc(8 * 4);
  float* c_mean = (float*)alloc(4096 * 4);
  float* c_stdv = (float*)alloc(4096 * 4);
  float* smM    = (float*)alloc(4096 * 4);
  float* ssM    = (float*)alloc(4096 * 4);
  float* sgM    = (float*)alloc(512 * 4);
  float* sgS    = (float*)alloc(512 * 4);
  float* scale  = (float*)alloc(4096 * 4);
  float* shift  = (float*)alloc(4096 * 4);

  auto blocks = [](long total) -> unsigned {
    long b = (total + 255) / 256;
    if (b > 65535) b = 65535;
    return (unsigned)b;
  };

  // weight repack+convert (f32 -> packed f16, L2 resident afterwards)
  k_repack_w<<<blocks(512L * 512 * 9), 256, 0, stream>>>(W11, wp11, 512, 512);
  k_repack_w<<<blocks(256L * 512 * 9), 256, 0, stream>>>(W12, wp12, 256, 512);
  k_repack_w<<<blocks(256L * 256 * 9), 256, 0, stream>>>(W21, wp21, 256, 256);
  k_repack_w<<<blocks(256L * 256 * 9), 256, 0, stream>>>(W22, wp22, 256, 256);
  k_repack_w<<<blocks(256L * 256 * 9), 256, 0, stream>>>(W23, wp23, 256, 256);
  k_repack_w<<<blocks(128L * 256 * 9), 256, 0, stream>>>(W24, wp24, 128, 256);
  k_repack_w<<<blocks(128L * 128 * 9), 256, 0, stream>>>(W31, wp31, 128, 128);
  k_repack_w<<<blocks(128L * 128 * 9), 256, 0, stream>>>(W32, wp32, 128, 128);
  k_repack_w<<<blocks(64L  * 128 * 9), 256, 0, stream>>>(W33, wp33, 64, 128);

  // f2 -> f16
  k_f32tof16<<<blocks(4L * 512 * 32 * 32), 256, 0, stream>>>(f2, bufA, 4L * 512 * 32 * 32);

  // conv chain (WMMA implicit GEMM; x-dim = H * W/32 pixel tiles)
  k_conv3x3_wmma<<<dim3(32 * 1, 4, 4), 256, 0, stream>>>(bufA, wp11, B11, bufB, 512, 512, 32, 32);
  k_conv3x3_wmma<<<dim3(32 * 1, 2, 4), 256, 0, stream>>>(bufB, wp12, B12, bufA, 512, 256, 32, 32);
  k_unpool_shift<<<blocks(4L * 256 * 64 * 64), 256, 0, stream>>>(bufA, s2lh, s2hl, s2hh, bufB, 4 * 256, 32, 32);
  k_conv3x3_wmma<<<dim3(64 * 2, 2, 4), 256, 0, stream>>>(bufB, wp21, B21, bufA, 256, 256, 64, 64);
  k_conv3x3_wmma<<<dim3(64 * 2, 2, 4), 256, 0, stream>>>(bufA, wp22, B22, bufB, 256, 256, 64, 64);
  k_conv3x3_wmma<<<dim3(64 * 2, 2, 4), 256, 0, stream>>>(bufB, wp23, B23, bufA, 256, 256, 64, 64);
  k_conv3x3_wmma<<<dim3(64 * 2, 1, 4), 256, 0, stream>>>(bufA, wp24, B24, bufB, 256, 128, 64, 64);
  k_unpool_shift<<<blocks(4L * 128 * 128 * 128), 256, 0, stream>>>(bufB, s1lh, s1hl, s1hh, bufA, 4 * 128, 64, 64);

  // AdaIN per class: content = bufA, style = f1
  k_pool_masks<<<blocks(8L * 128 * 128), 256, 0, stream>>>(cmask, cmp);
  k_pool_masks<<<blocks(8L * 128 * 128), 256, 0, stream>>>(smask, smp);
  k_mask_flags<<<dim3(8), 256, 0, stream>>>(cmask, smask, flags, sempty);
  k_stats<<<dim3(512), 256, 0, stream>>>(bufA, (const float*)nullptr, cmp,
                                         c_mean, c_stdv, (float*)nullptr, (float*)nullptr,
                                         4, 128, 128 * 128);
  k_stats<<<dim3(512), 256, 0, stream>>>((const _Float16*)nullptr, f1, smp,
                                         smM, ssM, sgM, sgS, 4, 128, 128 * 128);
  k_scale_shift<<<dim3(16), 256, 0, stream>>>(c_mean, c_stdv, smM, ssM, sgM, sgS,
                                              flags, sempty, scale, shift, 4, 128);
  k_adain_apply<<<blocks(4L * 128 * 128 * 128), 256, 0, stream>>>(bufA, cmp, scale, shift,
                                                                  bufB, 4, 128, 128 * 128);

  // final conv block
  k_conv3x3_wmma<<<dim3(128 * 4, 1, 4), 256, 0, stream>>>(bufB, wp31, B31, bufA, 128, 128, 128, 128);
  k_conv3x3_wmma<<<dim3(128 * 4, 1, 4), 256, 0, stream>>>(bufA, wp32, B32, bufB, 128, 128, 128, 128);
  k_conv3x3_wmma<<<dim3(128 * 4, 1, 4), 256, 0, stream>>>(bufB, wp33, B33, bufA, 128, 64, 128, 128);

  // final unpool + reflect pad -> f32 output (4,64,258,258)
  k_unpool_rpad<<<blocks(4L * 64 * 258 * 258), 256, 0, stream>>>(bufA, s0lh, s0hl, s0hh,
                                                                 (float*)d_out, 4 * 64, 128, 128);
}
